// Int8PhiAttention_61005715473018
// MI455X (gfx1250) — 16-bit and 8-bit matrix operand order repaired and checked by reading, compiled, not run on hardware
//
#include <hip/hip_runtime.h>
#include <cstdint>


typedef __attribute__((ext_vector_type(8)))  int       v8i;
typedef __attribute__((ext_vector_type(8)))  float     v8f;
typedef __attribute__((ext_vector_type(16))) _Float16  v16h;

constexpr int CB   = 2;
constexpr int CS   = 2048;
constexpr int CH   = 2560;
constexpr int CNH  = 32;
constexpr int CHD  = 80;
constexpr int CHDP = 96;
constexpr int CM   = CB * CS;
constexpr int KST  = CH / 64;

__device__ inline v8f zero8f() { v8f v; for (int i = 0; i < 8; ++i) v[i] = 0.f; return v; }
__device__ inline v8i zero8i() { v8i v; for (int i = 0; i < 8; ++i) v[i] = 0;   return v; }

__global__ __launch_bounds__(256) void k_quant(const float* __restrict__ x,
                                               int8_t* __restrict__ xq,
                                               const float* __restrict__ sp,
                                               long n) {
  long i = (long)blockIdx.x * 256 + threadIdx.x;
  if (i >= n) return;
  float s = sp[0];
  float v = rintf(x[i] / s);
  v = fminf(127.f, fmaxf(-128.f, v));
  xq[i] = (int8_t)v;
}

__global__ __launch_bounds__(256) void k_packw(const float* __restrict__ W,
                                               int* __restrict__ Wp) {
  long tid = (long)blockIdx.x * 256 + threadIdx.x;
  long total = (long)(CH / 16) * KST * 256;
  if (tid >= total) return;
  int  j    = (int)(tid & 7);
  int  lane = (int)((tid >> 3) & 31);
  long ft   = tid >> 8;
  int  kt   = (int)(ft % KST);
  long nt   = ft / KST;
  int  n    = (int)(nt * 16 + (lane & 15));
  int  kb   = kt * 64 + ((lane >> 4) & 1) * 8 + ((j >> 1) & 3) * 16 + (j & 1) * 4;
  const float* wr = W + (long)n * CH + kb;
  int pack = 0;
  for (int b = 0; b < 4; ++b) {
    int v = (int)lrintf(wr[b]);
    v = v < -128 ? -128 : (v > 127 ? 127 : v);
    pack |= (v & 0xFF) << (8 * b);
  }
  Wp[tid] = pack;
}

__global__ __launch_bounds__(256) void k_gemm_i8(const int8_t* __restrict__ Aq,
                                                 const int*    __restrict__ Wp,
                                                 const float*  __restrict__ bias,
                                                 const float*  __restrict__ sA,
                                                 const float*  __restrict__ sW,
                                                 float*        __restrict__ out) {
  __shared__ __align__(16) int8_t lA[128 * 64];
  const int tid  = threadIdx.x;
  const int wave = tid >> 5, lane = tid & 31;
  const int wm = wave >> 1, wn = wave & 1;
  const int col = lane & 15;
  const int hi8 = ((lane >> 4) & 1) * 8;
  const int n0 = blockIdx.x * 128;
  const int m0 = blockIdx.y * 128;

  v8i acc[2][4];
  for (int i = 0; i < 2; ++i)
    for (int j = 0; j < 4; ++j) acc[i][j] = zero8i();

  for (int kt = 0; kt < KST; ++kt) {
    __syncthreads();
    for (int i = 0; i < 2; ++i) {
      int off = tid * 16 + i * 4096;
      int row = off >> 6, c = off & 63;
      *(int4*)(lA + off) =
          *(const int4*)(Aq + (size_t)(m0 + row) * CH + (size_t)kt * 64 + c);
    }
    __syncthreads();

    v8i a[2];
    for (int fm = 0; fm < 2; ++fm) {
      int rA = wm * 32 + fm * 16 + col;
      const int8_t* base = lA + rA * 64 + hi8;
      long long d0 = *(const long long*)(base + 0);
      long long d1 = *(const long long*)(base + 16);
      long long d2 = *(const long long*)(base + 32);
      long long d3 = *(const long long*)(base + 48);
      v8i av;
      av[0] = (int)d0; av[1] = (int)(d0 >> 32);
      av[2] = (int)d1; av[3] = (int)(d1 >> 32);
      av[4] = (int)d2; av[5] = (int)(d2 >> 32);
      av[6] = (int)d3; av[7] = (int)(d3 >> 32);
      a[fm] = av;
    }

    for (int fn = 0; fn < 4; ++fn) {
      long ntile = (n0 + wn * 64 + fn * 16) >> 4;
      const int4* wp = (const int4*)(Wp + (((ntile * KST + kt) * 32 + lane) << 3));
      int4 b0 = wp[0], b1 = wp[1];
      v8i b;
      b[0] = b0.x; b[1] = b0.y; b[2] = b0.z; b[3] = b0.w;
      b[4] = b1.x; b[5] = b1.y; b[6] = b1.z; b[7] = b1.w;
      acc[0][fn] = __builtin_amdgcn_wmma_i32_16x16x64_iu8(true, a[0], true, b,
                                                          acc[0][fn], false, false);
      acc[1][fn] = __builtin_amdgcn_wmma_i32_16x16x64_iu8(true, a[1], true, b,
                                                          acc[1][fn], false, false);
    }
  }

  const float scale = sA[0] * sW[0];
  for (int fm = 0; fm < 2; ++fm)
    for (int fn = 0; fn < 4; ++fn) {
      int gn = n0 + wn * 64 + fn * 16 + col;
      float bb = bias[gn];
      for (int r = 0; r < 8; ++r) {
        int gm = m0 + wm * 32 + fm * 16 + hi8 + r;
        out[(size_t)gm * CH + gn] = (float)acc[fm][fn][r] * scale + bb;
      }
    }
}

__global__ __launch_bounds__(256) void k_ln_rope(const float* __restrict__ x,
                                                 const float* __restrict__ lw,
                                                 const float* __restrict__ lb,
                                                 _Float16* __restrict__ outh) {
  const int wave = threadIdx.x >> 5, lane = threadIdx.x & 31;
  long row = (long)blockIdx.x * 8 + wave;
  int  h  = (int)(row % CNH);
  long bs = row / CNH;
  int  s  = (int)(bs % CS);
  int  b  = (int)(bs / CS);

  const float* xr = x + row * CHD;
  float x0 = xr[lane];
  float x1 = xr[lane + 32];
  float x2 = (lane < 16) ? xr[lane + 64] : 0.f;

  float sum = x0 + x1 + x2;
  for (int o = 16; o; o >>= 1) sum += __shfl_xor(sum, o, 32);
  float mean = sum * (1.f / 80.f);
  float d0 = x0 - mean, d1 = x1 - mean, d2 = (lane < 16) ? (x2 - mean) : 0.f;
  float vs = d0 * d0 + d1 * d1 + d2 * d2;
  for (int o = 16; o; o >>= 1) vs += __shfl_xor(vs, o, 32);
  float inv = rsqrtf(vs * (1.f / 80.f) + 1e-5f);

  float y0 = d0 * inv * lw[lane] + lb[lane];
  float y1 = d1 * inv * lw[lane + 32] + lb[lane + 32];
  float y2 = (lane < 16) ? (d2 * inv * lw[lane + 64] + lb[lane + 64]) : 0.f;

  float part = __shfl_xor(y0, 16, 32);
  float rot  = (lane < 16) ? -part : part;
  float invf = __expf(-(float)(lane & 15) * (9.210340371976184f / 16.f));
  float ang  = (float)s * invf;
  y0 = y0 * cosf(ang) + rot * sinf(ang);

  _Float16* o = outh + (((long)b * CNH + h) * CS + s) * CHDP;
  o[lane]      = (_Float16)y0;
  o[lane + 32] = (_Float16)y1;
  if (lane < 16) o[lane + 64] = (_Float16)y2;
  else           o[64 + lane] = (_Float16)0.f;
}

__global__ __launch_bounds__(256) void k_vtrans(const float* __restrict__ v,
                                                _Float16* __restrict__ vt) {
  long i = (long)blockIdx.x * 256 + threadIdx.x;
  int  d  = (int)(i % CHD);
  long t  = i / CHD;
  int  h  = (int)(t % CNH);
  long bs = t / CNH;
  int  s  = (int)(bs % CS);
  int  b  = (int)(bs / CS);
  vt[(((long)b * CNH + h) * CHD + d) * CS + s] = (_Float16)v[i];
}

__global__ __launch_bounds__(256) void k_attn(const _Float16* __restrict__ qh,
                                              const _Float16* __restrict__ kh,
                                              const _Float16* __restrict__ vt,
                                              float* __restrict__ out) {
  __shared__ __align__(16) _Float16 lp[8][16 * 32];
  const int wave = threadIdx.x >> 5, lane = threadIdx.x & 31;
  const int col = lane & 15;
  const int hi  = (lane >> 4) & 1;
  const int b = blockIdx.z, h = blockIdx.y;
  const int qbase = blockIdx.x * 128 + wave * 16;
  const long bh = (long)b * CNH + h;

  const _Float16* q0 = qh + bh * CS * CHDP;
  const _Float16* k0 = kh + bh * CS * CHDP;
  const _Float16* v0 = vt + bh * CHD * (long)CS;
  _Float16* myp = lp[wave];

  v16h aq[3];
  {
    const _Float16* qr = q0 + (long)(qbase + col) * CHDP;
    for (int st = 0; st < 3; ++st) {
      union { int4 i[2]; v16h v; } u;
      u.i[0] = *(const int4*)(qr + st * 32 + hi * 16);
      u.i[1] = *(const int4*)(qr + st * 32 + hi * 16 + 8);
      aq[st] = u.v;
    }
  }

  v8f accO[5];
  for (int t = 0; t < 5; ++t) accO[t] = zero8f();
  float mrow[8], lrow[8];
  for (int r = 0; r < 8; ++r) { mrow[r] = -1e30f; lrow[r] = 0.f; }

  const int qend = qbase + 15;
  const float sscale = 0.11180339887498948f;

  for (int kb = 0; kb <= qend; kb += 32) {
    v8f s0 = zero8f(), s1 = zero8f();
    for (int st = 0; st < 3; ++st) {
      union { int4 i[2]; v16h v; } u0, u1;
      const _Float16* kr0 = k0 + (long)(kb + col) * CHDP + st * 32 + hi * 16;
      const _Float16* kr1 = k0 + (long)(kb + 16 + col) * CHDP + st * 32 + hi * 16;
      u0.i[0] = ((const int4*)kr0)[0]; u0.i[1] = ((const int4*)kr0)[1];
      u1.i[0] = ((const int4*)kr1)[0]; u1.i[1] = ((const int4*)kr1)[1];
      s0 = __builtin_amdgcn_wmma_f32_16x16x32_f16(false, aq[st], false, u0.v,
                                                  (short)0, s0, false, false);
      s1 = __builtin_amdgcn_wmma_f32_16x16x32_f16(false, aq[st], false, u1.v,
                                                  (short)0, s1, false, false);
    }

    float p0[8], p1[8];
    for (int r = 0; r < 8; ++r) {
      int q = qbase + hi * 8 + r;
      float a = s0[r] * sscale; if (kb + col > q)       a = -1e30f;
      float c = s1[r] * sscale; if (kb + 16 + col > q)  c = -1e30f;
      float mx = fmaxf(a, c);
      for (int o = 8; o; o >>= 1) mx = fmaxf(mx, __shfl_xor(mx, o, 32));
      float mnew = fmaxf(mrow[r], mx);
      float sc = __expf(mrow[r] - mnew);
      float e0 = __expf(a - mnew), e1 = __expf(c - mnew);
      float rs = e0 + e1;
      for (int o = 8; o; o >>= 1) rs += __shfl_xor(rs, o, 32);
      lrow[r] = lrow[r] * sc + rs;
      mrow[r] = mnew;
      for (int t = 0; t < 5; ++t) accO[t][r] *= sc;
      p0[r] = e0; p1[r] = e1;
    }

    for (int r = 0; r < 8; ++r) {
      int rr = hi * 8 + r;
      myp[rr * 32 + col]      = (_Float16)p0[r];
      myp[rr * 32 + 16 + col] = (_Float16)p1[r];
    }
    asm volatile("s_wait_dscnt 0" ::: "memory");
    union { int4 i[2]; v16h v; } up;
    {
      const _Float16* pr = myp + col * 32;
      up.i[0] = *(const int4*)(pr + hi * 16);
      up.i[1] = *(const int4*)(pr + hi * 16 + 8);
    }

    for (int t = 0; t < 5; ++t) {
      union { int4 i[2]; v16h v; } uv;
      const _Float16* vr = v0 + (long)(t * 16 + col) * CS + kb + hi * 16;
      uv.i[0] = ((const int4*)vr)[0];
      uv.i[1] = ((const int4*)vr)[1];
      accO[t] = __builtin_amdgcn_wmma_f32_16x16x32_f16(false, up.v, false, uv.v,
                                                       (short)0, accO[t], false, false);
    }
  }

  for (int t = 0; t < 5; ++t)
    for (int r = 0; r < 8; ++r) {
      int q = qbase + hi * 8 + r;
      out[((long)b * CS + q) * CH + h * CHD + t * 16 + col] = accO[t][r] / lrow[r];
    }
}

extern "C" void kernel_launch(void* const* d_in, const int* in_sizes, int n_in,
                              void* d_out, int out_size, void* d_ws, size_t ws_size,
                              hipStream_t stream) {
  (void)in_sizes; (void)n_in; (void)out_size; (void)ws_size;

  const float* hidden = (const float*)d_in[0];
  const float* W[4]   = { (const float*)d_in[1], (const float*)d_in[2],
                          (const float*)d_in[3], (const float*)d_in[4] };
  const float* bias[4]= { (const float*)d_in[5], (const float*)d_in[6],
                          (const float*)d_in[7], (const float*)d_in[8] };
  const float* s_attn  = (const float*)d_in[9];
  const float* s_wq    = (const float*)d_in[10];
  const float* s_wk    = (const float*)d_in[11];
  const float* s_wv    = (const float*)d_in[12];
  const float* s_dense = (const float*)d_in[13];
  const float* s_wo    = (const float*)d_in[14];
  const float* lnqw = (const float*)d_in[15];
  const float* lnqb = (const float*)d_in[16];
  const float* lnkw = (const float*)d_in[17];
  const float* lnkb = (const float*)d_in[18];

  char* ws = (char*)d_ws;
  size_t off = 0;
  auto carve = [&](size_t bytes) -> char* {
    char* p = ws + off;
    off += (bytes + 255) & ~(size_t)255;
    return p;
  };
  const size_t MH = (size_t)CM * CH;
  int8_t*   xq    = (int8_t*)carve(MH);
  int*      wp[4];
  for (int i = 0; i < 4; ++i) wp[i] = (int*)carve((size_t)CH * CH);
  float*    f32b  = (float*)carve(MH * 4);
  _Float16* qhb   = (_Float16*)carve((size_t)CB * CNH * CS * CHDP * 2);
  _Float16* khb   = (_Float16*)carve((size_t)CB * CNH * CS * CHDP * 2);
  _Float16* vtb   = (_Float16*)carve((size_t)CB * CNH * CHD * CS * 2);

  const int qblocks = (int)(MH / 256);
  const int pblocks = (int)(((size_t)(CH / 16) * KST * 256 + 255) / 256);
  dim3 ggrid(CH / 128, CM / 128);

  k_quant<<<qblocks, 256, 0, stream>>>(hidden, xq, s_attn, (long)MH);
  for (int i = 0; i < 4; ++i)
    k_packw<<<pblocks, 256, 0, stream>>>(W[i], wp[i]);

  k_gemm_i8<<<ggrid, 256, 0, stream>>>(xq, wp[0], bias[0], s_attn, s_wq, f32b);
  k_ln_rope<<<(CB * CS * CNH) / 8, 256, 0, stream>>>(f32b, lnqw, lnqb, qhb);

  k_gemm_i8<<<ggrid, 256, 0, stream>>>(xq, wp[1], bias[1], s_attn, s_wk, f32b);
  k_ln_rope<<<(CB * CS * CNH) / 8, 256, 0, stream>>>(f32b, lnkw, lnkb, khb);

  k_gemm_i8<<<ggrid, 256, 0, stream>>>(xq, wp[2], bias[2], s_attn, s_wv, f32b);
  k_vtrans<<<qblocks, 256, 0, stream>>>(f32b, vtb);

  k_attn<<<dim3(CS / 128, CNH, CB), 256, 0, stream>>>(qhb, khb, vtb, f32b);

  k_quant<<<qblocks, 256, 0, stream>>>(f32b, xq, s_dense, (long)MH);
  k_gemm_i8<<<ggrid, 256, 0, stream>>>(xq, wp[3], bias[3], s_dense, s_wo,
                                       (float*)d_out);
}
